// EdgeClassifierGNN_58085137711176
// MI455X (gfx1250) — matrix operand order repaired and checked by reading, not compiled, not run on hardware
//
#include <hip/hip_runtime.h>
#include <hip/hip_bf16.h>
#include <math.h>

#define NN 50000
#define NE 500000
#define HID 256
#define LAYERS 15
#define EPSV 1e-5f

typedef _Float16 half8  __attribute__((ext_vector_type(8)));
typedef _Float16 half16 __attribute__((ext_vector_type(16)));
typedef float    float8 __attribute__((ext_vector_type(8)));
typedef int      v4i    __attribute__((ext_vector_type(4)));

// CDNA5 async global->LDS DMA path (ASYNCcnt). Parameter types per hipcc
// diagnostic: (AS1 v4i*, AS3 v4i*, int offset, int cpol).
#if defined(__has_builtin)
#if __has_builtin(__builtin_amdgcn_global_load_async_to_lds_b128)
#define HAVE_ASYNC_LDS 1
#endif
#endif
#ifndef HAVE_ASYNC_LDS
#define HAVE_ASYNC_LDS 0
#endif

#if HAVE_ASYNC_LDS
typedef __attribute__((address_space(1))) v4i* gptr_v4i;
typedef __attribute__((address_space(3))) v4i* sptr_v4i;
#endif

static __device__ __forceinline__ void copy16_g2lds(const _Float16* gsrc, _Float16* ldst) {
#if HAVE_ASYNC_LDS
    __builtin_amdgcn_global_load_async_to_lds_b128((gptr_v4i)gsrc, (sptr_v4i)ldst, 0, 0);
#else
    *(half8*)ldst = *(const half8*)gsrc;
#endif
}

static __device__ __forceinline__ void async_fence_lds() {
#if HAVE_ASYNC_LDS
#if __has_builtin(__builtin_amdgcn_s_wait_asynccnt)
    __builtin_amdgcn_s_wait_asynccnt(0);
#else
    asm volatile("s_wait_asynccnt 0x0" ::: "memory");
#endif
#endif
}

// ---- WMMA fragment loaders (layouts per CDNA5 ISA §7.12.2) ----

// A-matrix 16x32 f16: lane l holds row (l&15); K = {kbase..+7, kbase+16..+23},
// kbase = (l<16) ? 0 : 8.  `row` points at this lane's LDS row.
static __device__ __forceinline__ half16 load_a_frag(const _Float16* row, int k0, int kbase) {
    half8 lo = *(const half8*)(row + k0 + kbase);
    half8 hi = *(const half8*)(row + k0 + kbase + 16);
    half16 a;
#pragma unroll
    for (int i = 0; i < 8; ++i) { a[i] = lo[i]; a[i + 8] = hi[i]; }
    return a;
}

// B-matrix 32x16 f16 from K-major (transposed) weight: lane l holds column N = n0+(l&15);
// K = {kbase..+7, kbase+16..+23}, kbase = (l<16) ? 0 : 8.  `col` points at this lane's column.
static __device__ __forceinline__ half16 load_b_frag(const _Float16* col, int k0, int kbase) {
    half16 b;
#pragma unroll
    for (int i = 0; i < 8; ++i) {
        b[i] = col[(k0 + kbase + i) * HID];
        b[i + 8] = col[(k0 + kbase + 16 + i) * HID];
    }
    return b;
}

// ---- one-time weight prep: f32 [out][in] -> f16 K-major [in][out] ----
__global__ void prep_weights_kernel(const float* __restrict__ Wrel,
                                    const float* __restrict__ Wroot,
                                    const float* __restrict__ W1,
                                    const float* __restrict__ W2,
                                    _Float16* __restrict__ Wrt,
                                    _Float16* __restrict__ Wot,
                                    _Float16* __restrict__ W1t,
                                    _Float16* __restrict__ W2t) {
    int tid = blockIdx.x * blockDim.x + threadIdx.x;
    int stride = gridDim.x * blockDim.x;
    const int rel_total = LAYERS * HID * HID;
    for (int i = tid; i < rel_total; i += stride) {
        int l = i >> 16;
        int r = i & 65535;
        int n = r >> 8;            // out channel
        int k = r & 255;           // in channel
        int t = l * HID * HID + k * HID + n;
        Wrt[t] = (_Float16)Wrel[i];
        Wot[t] = (_Float16)Wroot[i];
    }
    for (int i = tid; i < 2 * HID * HID; i += stride) {  // W1 [256][512]
        int n = i >> 9, k = i & 511;
        W1t[k * HID + n] = (_Float16)W1[i];
    }
    for (int i = tid; i < HID * HID; i += stride) {      // W2 [256][256]
        int n = i >> 8, k = i & 255;
        W2t[k * HID + n] = (_Float16)W2[i];
    }
}

// ---- h = x @ W_in^T  (IN_CH = 2); also writes f16 shadow ----
__global__ void input_fc_kernel(const float* __restrict__ x,
                                const float* __restrict__ W_in,
                                float* __restrict__ h,
                                _Float16* __restrict__ hf16) {
    int i = blockIdx.x * blockDim.x + threadIdx.x;
    if (i >= NN * HID) return;
    int n = i >> 8, c = i & 255;
    float v = x[n * 2 + 0] * W_in[c * 2 + 0] + x[n * 2 + 1] * W_in[c * 2 + 1];
    h[i] = v;
    hf16[i] = (_Float16)v;
}

// ---- zero agg + BN accumulators ----
__global__ void zero_kernel(float* __restrict__ agg,
                            float* __restrict__ bnsum,
                            float* __restrict__ bnsumsq) {
    int gid = blockIdx.x * blockDim.x + threadIdx.x;
    int stride = gridDim.x * blockDim.x;
    for (int i = gid; i < NN * HID; i += stride) agg[i] = 0.f;
    if (gid < HID) { bnsum[gid] = 0.f; bnsumsq[gid] = 0.f; }
}

// ---- segment_sum: agg[dst] += hf16[src]  (32 threads/edge, half8 per thread,
//      f32 atomic accumulation; agg stays L2-resident) ----
__global__ void scatter_add_kernel(const _Float16* __restrict__ hf16,
                                   const int* __restrict__ src,
                                   const int* __restrict__ dst,
                                   float* __restrict__ agg) {
    int gid = blockIdx.x * blockDim.x + threadIdx.x;
    int stride = gridDim.x * blockDim.x;
    const int total = NE * 32;
    for (int i = gid; i < total; i += stride) {
        int e = i >> 5;
        int c = (i & 31) * 8;
        int s = src[e], d = dst[e];
        half8 v = *(const half8*)(hf16 + s * HID + c);
        float* a = agg + d * HID + c;
#pragma unroll
        for (int j = 0; j < 8; ++j) atomicAdd(a + j, (float)v[j]);
    }
}

// ---- out = agg @ Wr^T + h @ Wo^T + br ; accumulate BN sum/sumsq ----
// 512 threads = 16 waves; block covers 64 rows x 256 cols; wave w owns cols
// [16w,16w+16) with 4 M-subtiles -> 4 independent WMMA accumulation chains and
// 4x B-fragment reuse.
__global__ __launch_bounds__(512) void gemm_layer_kernel(
        const float* __restrict__ agg, const _Float16* __restrict__ hf16,
        const _Float16* __restrict__ Wrt, const _Float16* __restrict__ Wot,
        const float* __restrict__ br,
        float* __restrict__ out, float* __restrict__ bnsum, float* __restrict__ bnsumsq) {
    __shared__ _Float16 sAgg[64 * HID];  // 32 KB
    __shared__ _Float16 sH[64 * HID];    // 32 KB
    const int m0 = blockIdx.x * 64;
    const int tid = threadIdx.x;

    // stage agg tile (f32 -> f16 convert)
    for (int i = tid; i < 64 * HID / 4; i += 512) {
        int r = i >> 6;
        int c4 = (i & 63) * 4;
        int row = m0 + r;
        float4 v = {0.f, 0.f, 0.f, 0.f};
        if (row < NN) v = *(const float4*)(agg + row * HID + c4);
        _Float16* d = sAgg + r * HID + c4;
        d[0] = (_Float16)v.x; d[1] = (_Float16)v.y;
        d[2] = (_Float16)v.z; d[3] = (_Float16)v.w;
    }
    // stage h tile: already f16 in global -> async DMA straight to LDS
    for (int i = tid; i < 64 * HID / 8; i += 512) {
        int r = i >> 5;
        int c8 = (i & 31) * 8;
        int row = m0 + r;
        if (row < NN) {
            copy16_g2lds(hf16 + row * HID + c8, sH + r * HID + c8);
        } else {
            half8 z = {};
            *(half8*)(sH + r * HID + c8) = z;
        }
    }
    async_fence_lds();
    __syncthreads();

    const int wave = tid >> 5;
    const int lane = tid & 31;
    const int n0 = wave * 16;
    const int mrow = lane & 15;
    const int kbase = (lane < 16) ? 0 : 8;

    float8 acc[4] = {};
#pragma unroll
    for (int k0 = 0; k0 < HID; k0 += 32) {
        if (k0 + 32 < HID) {  // global_prefetch_b8 on next weight K-slab
            __builtin_prefetch(Wrt + (k0 + 32 + lane) * HID + n0, 0, 1);
            __builtin_prefetch(Wot + (k0 + 32 + lane) * HID + n0, 0, 1);
        }
        half16 bR = load_b_frag(Wrt + n0 + (lane & 15), k0, kbase);
        half16 bO = load_b_frag(Wot + n0 + (lane & 15), k0, kbase);
#pragma unroll
        for (int s = 0; s < 4; ++s) {
            half16 aA = load_a_frag(sAgg + (s * 16 + mrow) * HID, k0, kbase);
            acc[s] = __builtin_amdgcn_wmma_f32_16x16x32_f16(false, aA, false, bR,
                                                            (short)0, acc[s], false, false);
            half16 aH = load_a_frag(sH + (s * 16 + mrow) * HID, k0, kbase);
            acc[s] = __builtin_amdgcn_wmma_f32_16x16x32_f16(false, aH, false, bO,
                                                            (short)0, acc[s], false, false);
        }
    }

    // C/D layout: col = n0 + (lane&15); rows = rbase + i, rbase = (lane<16)?0:8
    const int col = n0 + (lane & 15);
    const int rbase = (lane < 16) ? 0 : 8;
    const float bias = br[col];
    float ssum = 0.f, ssq = 0.f;
#pragma unroll
    for (int s = 0; s < 4; ++s) {
#pragma unroll
        for (int i = 0; i < 8; ++i) {
            int row = m0 + s * 16 + rbase + i;
            if (row < NN) {
                float v = acc[s][i] + bias;
                out[row * HID + col] = v;
                ssum += v;
                ssq += v * v;
            }
        }
    }
    // lanes l and l^16 hold the same column (different row halves)
    ssum += __shfl_xor(ssum, 16, 32);
    ssq  += __shfl_xor(ssq, 16, 32);
    if (lane < 16) {
        atomicAdd(&bnsum[col], ssum);
        atomicAdd(&bnsumsq[col], ssq);
    }
}

// ---- per-channel BN scale/shift ----
__global__ void bn_finalize_kernel(const float* __restrict__ bnsum,
                                   const float* __restrict__ bnsumsq,
                                   const float* __restrict__ g,
                                   const float* __restrict__ bt,
                                   float* __restrict__ scale,
                                   float* __restrict__ shift) {
    int c = threadIdx.x;
    const float inv_n = 1.f / (float)NN;
    float mean = bnsum[c] * inv_n;
    float var = bnsumsq[c] * inv_n - mean * mean;
    float sc = g[c] * rsqrtf(var + EPSV);
    scale[c] = sc;
    shift[c] = bt[c] - mean * sc;
}

// ---- h += relu(out * scale + shift); refresh f16 shadow ----
__global__ void update_h_kernel(float* __restrict__ h,
                                _Float16* __restrict__ hf16,
                                const float* __restrict__ out,
                                const float* __restrict__ scale,
                                const float* __restrict__ shift) {
    int i = blockIdx.x * blockDim.x + threadIdx.x;
    if (i >= NN * HID) return;
    int c = i & 255;
    float v = out[i] * scale[c] + shift[c];
    float hn = h[i] + (v > 0.f ? v : 0.f);
    h[i] = hn;
    hf16[i] = (_Float16)hn;
}

// ---- edge MLP: sigmoid(relu(relu(cat(h[s],h[d]) W1^T + b1) W2^T + b2) W3^T + b3) ----
// 512 threads = 16 waves; 32 edges per block; 2 M-subtiles per wave; z1/z2 in LDS.
__global__ __launch_bounds__(512) void edge_mlp_kernel(
        const _Float16* __restrict__ hf16,
        const int* __restrict__ src, const int* __restrict__ dst,
        const _Float16* __restrict__ W1t, const float* __restrict__ b1,
        const _Float16* __restrict__ W2t, const float* __restrict__ b2,
        const float* __restrict__ W3, const float* __restrict__ b3,
        float* __restrict__ out) {
    __shared__ _Float16 sE[32 * 512];   // 32 KB
    __shared__ _Float16 sZ1[32 * HID];  // 16 KB
    __shared__ _Float16 sZ2[32 * HID];  // 16 KB
    const int e0 = blockIdx.x * 32;
    const int tid = threadIdx.x;

    // gather cat(hf16[src], hf16[dst]) -> f16 tile [32][512] via async DMA
    for (int i = tid; i < 32 * 512 / 8; i += 512) {
        int r = i >> 6;             // / 64 chunks-per-row
        int c8 = (i & 63) * 8;      // 0..504
        int e = e0 + r;
        if (e < NE) {
            int node = (c8 < HID) ? src[e] : dst[e];
            copy16_g2lds(hf16 + node * HID + (c8 & 255), sE + r * 512 + c8);
        } else {
            half8 z = {};
            *(half8*)(sE + r * 512 + c8) = z;
        }
    }
    async_fence_lds();
    __syncthreads();

    const int wave = tid >> 5;
    const int lane = tid & 31;
    const int n0 = wave * 16;
    const int mrow = lane & 15;
    const int kbase = (lane < 16) ? 0 : 8;
    const int col = n0 + (lane & 15);
    const int rbase = (lane < 16) ? 0 : 8;

    // z1 = relu(E @ W1^T + b1), K = 512
    float8 acc1[2] = {};
#pragma unroll
    for (int k0 = 0; k0 < 512; k0 += 32) {
        if (k0 + 32 < 512)
            __builtin_prefetch(W1t + (k0 + 32 + lane) * HID + n0, 0, 1);
        half16 b = load_b_frag(W1t + n0 + (lane & 15), k0, kbase);
#pragma unroll
        for (int s = 0; s < 2; ++s) {
            half16 a = load_a_frag(sE + (s * 16 + mrow) * 512, k0, kbase);
            acc1[s] = __builtin_amdgcn_wmma_f32_16x16x32_f16(false, a, false, b,
                                                             (short)0, acc1[s], false, false);
        }
    }
    {
        float bias = b1[col];
#pragma unroll
        for (int s = 0; s < 2; ++s)
#pragma unroll
            for (int i = 0; i < 8; ++i) {
                float v = acc1[s][i] + bias;
                sZ1[(s * 16 + rbase + i) * HID + col] = (_Float16)(v > 0.f ? v : 0.f);
            }
    }
    __syncthreads();

    // z2 = relu(z1 @ W2^T + b2), K = 256
    float8 acc2[2] = {};
#pragma unroll
    for (int k0 = 0; k0 < HID; k0 += 32) {
        if (k0 + 32 < HID)
            __builtin_prefetch(W2t + (k0 + 32 + lane) * HID + n0, 0, 1);
        half16 b = load_b_frag(W2t + n0 + (lane & 15), k0, kbase);
#pragma unroll
        for (int s = 0; s < 2; ++s) {
            half16 a = load_a_frag(sZ1 + (s * 16 + mrow) * HID, k0, kbase);
            acc2[s] = __builtin_amdgcn_wmma_f32_16x16x32_f16(false, a, false, b,
                                                             (short)0, acc2[s], false, false);
        }
    }
    {
        float bias = b2[col];
#pragma unroll
        for (int s = 0; s < 2; ++s)
#pragma unroll
            for (int i = 0; i < 8; ++i) {
                float v = acc2[s][i] + bias;
                sZ2[(s * 16 + rbase + i) * HID + col] = (_Float16)(v > 0.f ? v : 0.f);
            }
    }
    __syncthreads();

    // final dot + sigmoid: wave w handles edges e0 + 2w, e0 + 2w + 1
#pragma unroll
    for (int j = 0; j < 2; ++j) {
        int r = wave * 2 + j;
        int e = e0 + r;
        if (e < NE) {
            const _Float16* zrow = sZ2 + r * HID;
            float t = 0.f;
#pragma unroll
            for (int q = 0; q < 8; ++q) {
                int c = lane * 8 + q;
                t += (float)zrow[c] * W3[c];
            }
#pragma unroll
            for (int off = 16; off > 0; off >>= 1) t += __shfl_xor(t, off, 32);
            if (lane == 0) out[e] = 1.f / (1.f + expf(-(t + b3[0])));
        }
    }
}

extern "C" void kernel_launch(void* const* d_in, const int* in_sizes, int n_in,
                              void* d_out, int out_size, void* d_ws, size_t ws_size,
                              hipStream_t stream) {
    (void)in_sizes; (void)n_in; (void)out_size; (void)ws_size;
    const float* x     = (const float*)d_in[0];
    const int*   ei    = (const int*)d_in[1];
    const float* W_in  = (const float*)d_in[2];
    const float* Wrel  = (const float*)d_in[3];
    const float* brel  = (const float*)d_in[4];
    const float* Wroot = (const float*)d_in[5];
    const float* gamma = (const float*)d_in[6];
    const float* beta  = (const float*)d_in[7];
    const float* W1    = (const float*)d_in[8];
    const float* b1    = (const float*)d_in[9];
    const float* W2    = (const float*)d_in[10];
    const float* b2    = (const float*)d_in[11];
    const float* W3    = (const float*)d_in[12];
    const float* b3    = (const float*)d_in[13];
    const int* src = ei;
    const int* dst = ei + NE;

    // workspace carve-out (256B aligned slabs)
    char* p = (char*)d_ws;
    auto alloc = [&](size_t bytes) -> void* {
        void* r = (void*)p;
        p += (bytes + 255) & ~(size_t)255;
        return r;
    };
    float*    h       = (float*)alloc(sizeof(float) * NN * HID);
    _Float16* hf16    = (_Float16*)alloc(sizeof(_Float16) * NN * HID);
    float*    agg     = (float*)alloc(sizeof(float) * NN * HID);
    float*    outb    = (float*)alloc(sizeof(float) * NN * HID);
    float*    bnsum   = (float*)alloc(sizeof(float) * HID);
    float*    bnsumsq = (float*)alloc(sizeof(float) * HID);
    float*    bnscale = (float*)alloc(sizeof(float) * HID);
    float*    bnshift = (float*)alloc(sizeof(float) * HID);
    _Float16* Wrt     = (_Float16*)alloc(sizeof(_Float16) * LAYERS * HID * HID);
    _Float16* Wot     = (_Float16*)alloc(sizeof(_Float16) * LAYERS * HID * HID);
    _Float16* W1t     = (_Float16*)alloc(sizeof(_Float16) * 2 * HID * HID);
    _Float16* W2t     = (_Float16*)alloc(sizeof(_Float16) * HID * HID);

    prep_weights_kernel<<<512, 256, 0, stream>>>(Wrel, Wroot, W1, W2, Wrt, Wot, W1t, W2t);
    input_fc_kernel<<<(NN * HID + 255) / 256, 256, 0, stream>>>(x, W_in, h, hf16);

    for (int l = 0; l < LAYERS; ++l) {
        zero_kernel<<<2048, 256, 0, stream>>>(agg, bnsum, bnsumsq);
        scatter_add_kernel<<<4096, 256, 0, stream>>>(hf16, src, dst, agg);
        gemm_layer_kernel<<<(NN + 63) / 64, 512, 0, stream>>>(
            agg, hf16, Wrt + l * HID * HID, Wot + l * HID * HID, brel + l * HID,
            outb, bnsum, bnsumsq);
        bn_finalize_kernel<<<1, HID, 0, stream>>>(
            bnsum, bnsumsq, gamma + l * HID, beta + l * HID, bnscale, bnshift);
        update_h_kernel<<<(NN * HID + 255) / 256, 256, 0, stream>>>(
            h, hf16, outb, bnscale, bnshift);
    }

    edge_mlp_kernel<<<(NE + 31) / 32, 512, 0, stream>>>(
        hf16, src, dst, W1t, b1, W2t, b2, W3, b3, (float*)d_out);
}
